// SMoEGateNet_36361193128717
// MI455X (gfx1250) — matrix operand order repaired and checked by reading, not compiled, not run on hardware
//
#include <hip/hip_runtime.h>
#include <math.h>

// ---------------------------------------------------------------------------
// SMoE gate net for MI455X (gfx1250, wave32, WMMA).
//   WG = 64 tokens (4 strips x 16). Each wave owns (strip, N-half):
//     half0: experts 0-3 h-tiles + gate tile ; half1: experts 4-7 h-tiles.
//   __launch_bounds__(256) lifts the 128-VGPR cap -> accumulators resident.
//   Pipelining: B-fragments in a rotating 4-deep register buffer with
//   compile-time indices (no copies for the allocator to coalesce), so each
//   tile's LDS load is issued 4 WMMAs ahead; next chunk's x loads issued
//   before the WMMA block. Weights staged to LDS via TDM double buffering;
//   softmax/top2 in fp32; combine folded into h before layer-2 WMMA.
// ---------------------------------------------------------------------------

typedef __attribute__((ext_vector_type(16))) __bf16        v16bf;
typedef __attribute__((ext_vector_type(8)))  float         v8f;
typedef __attribute__((ext_vector_type(4)))  float         v4f;
typedef __attribute__((ext_vector_type(4)))  unsigned int  u32x4;
typedef __attribute__((ext_vector_type(8)))  int           i32x8;
typedef __attribute__((ext_vector_type(4)))  int           i32x4;

#define BTOK   131072
#define INDIM  512
#define MKTD   16
#define NEXP   8
#define EXPD   64
#define OUTD   32
#define GIN    528                       // IN_DIM + MKT_DIM
#define NTOT   528                       // 8*64 expert cols + 8 gate cols + 8 pad
#define KCH    17                        // 16 chunks of x (K=512) + 1 market/gate tail
#define CHUNK_ELEMS (NTOT * 32)          // 16896 bf16 per K-chunk of packed W
#define W1F_ELEMS   (KCH * CHUNK_ELEMS)  // 287232
#define W2F_ELEMS   (NEXP * 2 * 32 * 32) // 16384

// LDS layout (bytes)
#define SW_OFF   0
#define SW_BYTES (2 * CHUNK_ELEMS * 2)       // 67584 double-buffered W chunks
#define SH_OFF   (SW_OFF + SW_BYTES)
#define SH_BYTES (8 * 16 * 72 * 2)           // 18432 per-wave h bounce
#define SL_OFF   (SH_OFF + SH_BYTES)
#define SL_BYTES (4 * 256 * 4)               // 4096 per-strip gate logits
#define SC_OFF   (SL_OFF + SL_BYTES)
#define SC_BYTES (4 * 128 * 4)               // 2048 per-strip combine weights
#define SO_OFF   (SC_OFF + SC_BYTES)
#define SO_BYTES (4 * 16 * 32 * 4)           // 8192 per-strip half1 partial out
#define SMEM_BYTES (SO_OFF + SO_BYTES)       // 100352

#if __has_builtin(__builtin_amdgcn_tensor_load_to_lds) && __has_builtin(__builtin_amdgcn_s_wait_tensorcnt)
#define USE_TDM 1
#else
#define USE_TDM 0
#endif

struct Frag32 { u32x4 a, b; };               // 32B register fragment (8 VGPRs)

__device__ __forceinline__ unsigned short f2bf(float f) {
  unsigned int u = __builtin_bit_cast(unsigned int, f);
  unsigned int r = u + 0x7FFFu + ((u >> 16) & 1u);   // round-to-nearest-even
  return (unsigned short)(r >> 16);
}
#if __has_builtin(__builtin_amdgcn_cvt_pk_bf16_f32)
__device__ __forceinline__ unsigned int pk2(float a, float b) {
  auto r = __builtin_amdgcn_cvt_pk_bf16_f32(a, b);   // one VALU op per pair
  return __builtin_bit_cast(unsigned int, r);
}
#else
__device__ __forceinline__ unsigned int pk2(float a, float b) {
  return (unsigned int)f2bf(a) | ((unsigned int)f2bf(b) << 16);
}
#endif
__device__ __forceinline__ v16bf ld_bfrag(const unsigned short* p) {
  Frag32 f;
  f.a = *(const u32x4*)p;
  f.b = *(const u32x4*)(p + 16);
  return __builtin_bit_cast(v16bf, f);
}

// --- TDM: 1D copy of one 33792B weight chunk (data_size=8B, tile_dim0=4224) ---
__device__ __forceinline__ void tdm_issue(const unsigned short* gsrc, unsigned short* ldst) {
#if USE_TDM
  unsigned long long ga = (unsigned long long)(const void*)gsrc;
  unsigned int la = (unsigned int)(unsigned long long)(void*)ldst;  // LDS byte offset
  u32x4 g0;
  g0.x = 1u;                                            // count=1, is_restore=0, gather off
  g0.y = la;                                            // lds_addr
  g0.z = (unsigned int)(ga & 0xffffffffull);            // global_addr[31:0]
  g0.w = (unsigned int)((ga >> 32) & 0x01ffffffull) | 0x80000000u;  // addr[56:32] | type=2
  i32x8 g1;
  g1[0] = 0x00030000;                                   // data_size=3 (8 bytes)
  g1[1] = (int)(4224u << 16);                           // tensor_dim0 = 4224
  g1[2] = 0x00010000;                                   // tensor_dim1 = 1
  g1[3] = (int)(4224u << 16);                           // tile_dim0 = 4224
  g1[4] = 1;                                            // tile_dim1 = 1
  g1[5] = 4224;                                         // tensor_dim0_stride
  g1[6] = 0; g1[7] = 0;
  i32x4 gz4 = {0, 0, 0, 0};                             // groups 2/3 unused (<=2D tensor)
  i32x8 gz8 = {0, 0, 0, 0, 0, 0, 0, 0};
  __builtin_amdgcn_tensor_load_to_lds(g0, g1, gz4, gz4, gz8, 0);
#else
  (void)gsrc; (void)ldst;
#endif
}

// ---------------------------------------------------------------------------
// Pack w1|gate_w and w2 (f32) into bf16 WMMA B-fragment layout in workspace.
//   w1f element ((c*528+n)*32 + t) holds W[k=c*32+t][n] where
//   n<512 -> w1[e=n/64][k][n%64]; 512<=n<520 -> gate_w[k][n-512]; else 0.
//   w2f element (((e*2+c)*32+n)*32 + t) holds w2[e][k=c*32+t][n].
// ---------------------------------------------------------------------------
__global__ void __launch_bounds__(256)
pack_weights_kernel(const float* __restrict__ w1,
                    const float* __restrict__ gate_w,
                    const float* __restrict__ w2,
                    unsigned short* __restrict__ w1f,
                    unsigned short* __restrict__ w2f) {
  int idx = blockIdx.x * blockDim.x + threadIdx.x;
  if (idx < W1F_ELEMS) {
    int c = idx / CHUNK_ELEMS;
    int rem = idx - c * CHUNK_ELEMS;
    int n = rem >> 5;
    int t = rem & 31;
    int k = c * 32 + t;
    float v = 0.f;
    if (n < 512) {
      if (k < INDIM) v = w1[((n >> 6) * INDIM + k) * EXPD + (n & 63)];
    } else if (n < 520) {
      if (k < GIN) v = gate_w[k * NEXP + (n - 512)];
    }
    w1f[idx] = f2bf(v);
  } else if (idx < W1F_ELEMS + W2F_ELEMS) {
    int j = idx - W1F_ELEMS;
    int e = j >> 11;
    int rem = j & 2047;
    int c2 = rem >> 10;
    int rem2 = rem & 1023;
    int n = rem2 >> 5;
    int t = rem2 & 31;
    int k = c2 * 32 + t;
    w2f[j] = f2bf(w2[(e * EXPD + k) * OUTD + n]);
  }
}

// ---------------------------------------------------------------------------
// Main fused kernel: 256 threads = 8 waves = 4 strips x 2 N-halves.
// ---------------------------------------------------------------------------
__global__ void __launch_bounds__(256)
moe_fused_kernel(const float* __restrict__ x,
                 const float* __restrict__ mkt,
                 const float* __restrict__ gate_b,
                 const float* __restrict__ b1,
                 const float* __restrict__ b2,
                 const unsigned short* __restrict__ w1f,
                 const unsigned short* __restrict__ w2f,
                 float* __restrict__ out,
                 float* __restrict__ gout) {
  extern __shared__ __align__(16) unsigned char smem[];
  unsigned short* sW = (unsigned short*)(smem + SW_OFF);
  unsigned short* sH = (unsigned short*)(smem + SH_OFF);
  float* sLogit = (float*)(smem + SL_OFF);
  float* sComb  = (float*)(smem + SC_OFF);
  float* sOut   = (float*)(smem + SO_OFF);

  const int tid   = threadIdx.x;
  const int lane  = tid & 31;
  const int wv    = tid >> 5;
  const int strip = wv >> 1;       // 0..3 : which 16-token strip
  const int nhalf = wv & 1;        // 0: experts 0-3 + gate ; 1: experts 4-7
  const int half  = lane >> 4;     // 0: lanes 0-15, 1: lanes 16-31
  const int l16   = lane & 15;
  const int rowBase = blockIdx.x * 64 + strip * 16;
  const int myrow   = rowBase + l16;
  const int tbase   = nhalf * 16;  // first owned N-tile

  const float* xrow = x + (size_t)myrow * INDIM;
  const float* mrow = mkt + (size_t)myrow * MKTD;

  // ---- stage chunk 0
#if USE_TDM
  if (wv == 0) tdm_issue(w1f, sW);
#else
  for (int i = tid; i < CHUNK_ELEMS / 8; i += 256)
    *(u32x4*)(sW + i * 8) = *(const u32x4*)(w1f + i * 8);
#endif

  v8f vzero = {0.f, 0.f, 0.f, 0.f, 0.f, 0.f, 0.f, 0.f};
  v4f fzero = {0.f, 0.f, 0.f, 0.f};
  v8f acc[16];                     // 4 experts x 4 N-tiles
  v8f accG = vzero;                // gate tile (half0 only)
#pragma unroll
  for (int t = 0; t < 16; ++t) acc[t] = vzero;

  // A-source prefetch registers (chunk c data lives in xf during iteration c)
  v4f xf[4], xfn[4];
  {
    const float* px = xrow + half * 8;               // chunk 0
    xf[0] = *(const v4f*)(px);      xf[1] = *(const v4f*)(px + 4);
    xf[2] = *(const v4f*)(px + 16); xf[3] = *(const v4f*)(px + 20);
  }

  // ---- fused K-loop: gate logits + layer-1 share each A-fragment
  for (int c = 0; c < KCH; ++c) {
    if (c + 1 < KCH) {
#if USE_TDM
      if (wv == 0) {
        tdm_issue(w1f + (size_t)(c + 1) * CHUNK_ELEMS, sW + ((c + 1) & 1) * CHUNK_ELEMS);
        __builtin_amdgcn_s_wait_tensorcnt(1);   // chunk c landed, chunk c+1 in flight
      }
#else
      for (int i = tid; i < CHUNK_ELEMS / 8; i += 256)
        *(u32x4*)(sW + ((c + 1) & 1) * CHUNK_ELEMS + i * 8) =
            *(const u32x4*)(w1f + (size_t)(c + 1) * CHUNK_ELEMS + i * 8);
#endif
    } else {
#if USE_TDM
      if (wv == 0) __builtin_amdgcn_s_wait_tensorcnt(0);
#endif
    }
    __syncthreads();

    // A-fragment (16x32 bf16): lane<16 holds K {0..7,16..23}, lane>=16 {8..15,24..31}
    // (market tail packs zeros in its upper half, so the build is uniform)
    Frag32 a;
    a.a.x = pk2(xf[0].x, xf[0].y); a.a.y = pk2(xf[0].z, xf[0].w);
    a.a.z = pk2(xf[1].x, xf[1].y); a.a.w = pk2(xf[1].z, xf[1].w);
    a.b.x = pk2(xf[2].x, xf[2].y); a.b.y = pk2(xf[2].z, xf[2].w);
    a.b.z = pk2(xf[3].x, xf[3].y); a.b.w = pk2(xf[3].z, xf[3].w);
    v16bf A = __builtin_bit_cast(v16bf, a);

    // prefetch next chunk's A-source: latency hidden behind the WMMA block
    if (c + 1 < 16) {
      const float* px = xrow + (c + 1) * 32 + half * 8;
      xfn[0] = *(const v4f*)(px);      xfn[1] = *(const v4f*)(px + 4);
      xfn[2] = *(const v4f*)(px + 16); xfn[3] = *(const v4f*)(px + 20);
    } else if (c + 1 == 16) {
      const float* pm = mrow + half * 8;               // market tail: K 512..527
      xfn[0] = *(const v4f*)(pm);      xfn[1] = *(const v4f*)(pm + 4);
      xfn[2] = fzero;                  xfn[3] = fzero;
    }

    const unsigned short* wb = sW + (c & 1) * CHUNK_ELEMS;
    if (c < 16) {
      const unsigned short* bbase = wb + ((tbase * 16 + l16) * 32 + half * 8);
      // rotating 4-deep B-fragment pipeline; indices are compile-time after
      // unrolling, so no register copies exist for the allocator to collapse
      Frag32 bf[4];
#pragma unroll
      for (int i = 0; i < 4; ++i) {
        const unsigned short* bp = bbase + i * 512;
        bf[i].a = *(const u32x4*)(bp);
        bf[i].b = *(const u32x4*)(bp + 16);
      }
#pragma unroll
      for (int tt = 0; tt < 16; ++tt) {
        acc[tt] = __builtin_amdgcn_wmma_f32_16x16x32_bf16(
            false, A, false, __builtin_bit_cast(v16bf, bf[tt & 3]), (short)0,
            acc[tt], false, false);
        if (tt + 4 < 16) {
          const unsigned short* bp = bbase + (tt + 4) * 512;
          bf[tt & 3].a = *(const u32x4*)(bp);
          bf[tt & 3].b = *(const u32x4*)(bp + 16);
        }
      }
    }
    if (nhalf == 0) {                // gate tile: all 17 chunks
      const unsigned short* bp = wb + ((32 * 16 + l16) * 32 + half * 8);
      v16bf Bm = ld_bfrag(bp);
      accG = __builtin_amdgcn_wmma_f32_16x16x32_bf16(
          false, A, false, Bm, (short)0, accG, false, false);
    }

    xf[0] = xfn[0]; xf[1] = xfn[1]; xf[2] = xfn[2]; xf[3] = xfn[3];
    __syncthreads();
  }

  // ---- gate (half0 wave per strip, wave-local LDS): softmax/top2/renorm
  float* Cw = sComb + strip * 128;               // combine[e][row], e-major
  if (nhalf == 0) {
    float* Lg = sLogit + strip * 256;
#pragma unroll
    for (int r = 0; r < 8; ++r)
      Lg[(half * 8 + r) * 16 + l16] = accG[r];
    if (lane < 16) {
      const int row = lane;
      float l[8], p[8];
      float m = -1e30f;
#pragma unroll
      for (int e = 0; e < 8; ++e) { l[e] = Lg[row * 16 + e] + gate_b[e]; m = fmaxf(m, l[e]); }
      float s = 0.f;
#pragma unroll
      for (int e = 0; e < 8; ++e) { p[e] = expf(l[e] - m); s += p[e]; }
      float inv = 1.f / s;
#pragma unroll
      for (int e = 0; e < 8; ++e) {
        p[e] *= inv;
        gout[(size_t)(rowBase + row) * NEXP + e] = p[e];
      }
      int i1 = 0;
#pragma unroll
      for (int e = 1; e < 8; ++e) if (p[e] > p[i1]) i1 = e;
      int i2 = -1; float v2 = -1.f;
#pragma unroll
      for (int e = 0; e < 8; ++e) if (e != i1 && p[e] > v2) { v2 = p[e]; i2 = e; }
      float v1 = p[i1];
      float invt = 1.f / (v1 + v2 + 1e-8f);
#pragma unroll
      for (int e = 0; e < 8; ++e)
        Cw[e * 16 + row] = (e == i1) ? v1 * invt : ((e == i2) ? v2 * invt : 0.f);
    }
  }
  __syncthreads();                               // publish Cw to half1 waves

  // ---- layer 2 (wave-local): bias+GELU+combine-scale h -> LDS -> WMMA
  v8f oacc0 = vzero, oacc1 = vzero;
  unsigned short* Hm = sH + wv * (16 * 72);      // [16 rows][72] bf16, padded stride

#pragma unroll
  for (int el = 0; el < 4; ++el) {
    const int e = tbase / 4 + el;                // global expert id (uniform per wave)
#pragma unroll
    for (int j = 0; j < 4; ++j) {
      const int col = j * 16 + l16;              // 0..63 within expert
      const float bias = b1[e * EXPD + col];
#pragma unroll
      for (int r = 0; r < 8; ++r) {
        const int row = half * 8 + r;
        float v = acc[el * 4 + j][r] + bias;
        v = 0.5f * v * (1.0f + erff(v * 0.70710678118654752f));   // exact GELU
        v *= Cw[e * 16 + row];                   // fold combine weight into h
        Hm[row * 72 + col] = f2bf(v);
      }
    }
    // wave-local h bounce: same-wave DS ordering, no block barrier needed
#pragma unroll
    for (int c2 = 0; c2 < 2; ++c2) {
      const unsigned short* ap = Hm + l16 * 72 + c2 * 32 + half * 8;
      Frag32 af;
      af.a = *(const u32x4*)ap;
      af.b = *(const u32x4*)(ap + 16);
      v16bf A2 = __builtin_bit_cast(v16bf, af);
#pragma unroll
      for (int nn = 0; nn < 2; ++nn) {
        const unsigned short* bp =
            w2f + (((e * 2 + c2) * 32 + nn * 16 + l16) * 32 + half * 8);
        v16bf B2 = ld_bfrag(bp);
        if (nn == 0)
          oacc0 = __builtin_amdgcn_wmma_f32_16x16x32_bf16(
              false, A2, false, B2, (short)0, oacc0, false, false);
        else
          oacc1 = __builtin_amdgcn_wmma_f32_16x16x32_bf16(
              false, A2, false, B2, (short)0, oacc1, false, false);
      }
    }
  }

  // ---- reduce the two N-halves per strip and store
  float* So = sOut + strip * (16 * 32);
  if (nhalf == 1) {
#pragma unroll
    for (int nn = 0; nn < 2; ++nn) {
      v8f oa = nn ? oacc1 : oacc0;
      const int col = nn * 16 + l16;
#pragma unroll
      for (int r = 0; r < 8; ++r)
        So[(half * 8 + r) * 32 + col] = oa[r];
    }
  }
  __syncthreads();
  if (nhalf == 0) {
#pragma unroll
    for (int nn = 0; nn < 2; ++nn) {
      v8f oa = nn ? oacc1 : oacc0;
      const int col = nn * 16 + l16;
      float bb[8];
#pragma unroll
      for (int e = 0; e < 8; ++e) bb[e] = b2[e * OUTD + col];
#pragma unroll
      for (int r = 0; r < 8; ++r) {
        const int row = half * 8 + r;
        float v = oa[r] + So[row * 32 + col];
#pragma unroll
        for (int e = 0; e < 8; ++e) v += Cw[e * 16 + row] * bb[e];
        out[(size_t)(rowBase + row) * OUTD + col] = v;
      }
    }
  }
}

extern "C" void kernel_launch(void* const* d_in, const int* in_sizes, int n_in,
                              void* d_out, int out_size, void* d_ws, size_t ws_size,
                              hipStream_t stream) {
  const float* x      = (const float*)d_in[0];
  const float* mkt    = (const float*)d_in[1];
  const float* gate_w = (const float*)d_in[2];
  const float* gate_b = (const float*)d_in[3];
  const float* w1     = (const float*)d_in[4];
  const float* b1     = (const float*)d_in[5];
  const float* w2     = (const float*)d_in[6];
  const float* b2     = (const float*)d_in[7];

  unsigned short* w1f = (unsigned short*)d_ws;               // 574464 B
  unsigned short* w2f = w1f + W1F_ELEMS;                     // + 32768 B

  float* out  = (float*)d_out;                               // [B,32]
  float* gout = out + (size_t)BTOK * OUTD;                   // [B,8] gate_probs

  const int packThreads = W1F_ELEMS + W2F_ELEMS;             // 303616
  pack_weights_kernel<<<(packThreads + 255) / 256, 256, 0, stream>>>(w1, gate_w, w2, w1f, w2f);

  moe_fused_kernel<<<dim3(BTOK / 64), dim3(256), SMEM_BYTES, stream>>>(
      x, mkt, gate_b, b1, b2, w1f, w2f, out, gout);

  (void)in_sizes; (void)n_in; (void)out_size; (void)ws_size;
}
